// DeepSeekV3MoE_66915590472170
// MI455X (gfx1250) — matrix operand order repaired and checked by reading, not compiled, not run on hardware
//
#include <hip/hip_runtime.h>
#include <hip/hip_bf16.h>

// ---------------- problem constants (fixed by reference) ----------------
constexpr int T_  = 2048;   // tokens
constexpr int D_  = 2048;   // hidden
constexpr int M_  = 1408;   // moe intermediate
constexpr int E_  = 32;     // experts
constexpr int K_  = 4;      // top-k
constexpr int MS_ = 2816;   // shared intermediate (M * 2)
constexpr int CAP_ = 512;   // capacity per expert
constexpr int TK_ = T_ * K_;
constexpr float ROUTE_SCALE_ = 2.5f;

// ---------------- WMMA types ----------------
typedef __attribute__((ext_vector_type(16))) __bf16 v16bf;
typedef __attribute__((ext_vector_type(8)))  float  v8f;

union AFrag { v16bf v; unsigned u[8]; };
union BFrag { v16bf v; unsigned u[8]; };

__device__ __forceinline__ v8f wmma_bf16(const AFrag& a, const BFrag& b, v8f c) {
    return __builtin_amdgcn_wmma_f32_16x16x32_bf16(
        /*neg_a=*/false, a.v, /*neg_b=*/false, b.v,
        /*c_mod=*/(short)0, c, /*reuse_a=*/false, /*reuse_b=*/false);
}

// ---------------- tiling ----------------
constexpr int BM = 128;        // rows per block
constexpr int BN = 64;         // cols per block
constexpr int BK = 32;         // k per stage (== WMMA K)
constexpr int LDA  = BK + 2;   // padded LDS stride; TDM pad (16dw interval, 1dw) matches
constexpr int LDBT = BK + 2;

// ---------------- Tensor Data Mover (TDM) helper -----------------------
// Loads a BM x BK bf16 tile (row stride = kdim elements) from global into
// LDS at lds_byte_off with one extra DWORD of padding per 64B row
// (pad_interval=3 -> 16 DWORDs, pad_amount=0 -> 1 DWORD) == LDA of 34 bf16.
// Descriptor packing per CDNA5 ISA section 8.3 / 8.4 (D# groups 0 and 1).
// This toolchain's builtin takes 6 args: (g0, g1, g2, g3, g_extra, cpol).
typedef unsigned int u32x4 __attribute__((ext_vector_type(4)));
typedef int          i32x8 __attribute__((ext_vector_type(8)));
typedef int          i32x4 __attribute__((ext_vector_type(4)));

__device__ __forceinline__ unsigned rfl(unsigned v) {
    return (unsigned)__builtin_amdgcn_readfirstlane((int)v);
}

__device__ __forceinline__ void tdm_load_a_tile(unsigned lds_byte_off,
                                                const __bf16* gptr,
                                                int kdim_elems,      // row stride
                                                int kdim_rem,        // tensor_dim0 (OOB)
                                                int rows_rem) {      // tensor_dim1 (OOB)
    unsigned long long ga = (unsigned long long)(uintptr_t)gptr;
    // ---- group 0: count=1, lds_addr, global_addr[56:0], type=2 ----
    unsigned g0d0 = rfl(1u);                                   // count=1, user D#
    unsigned g0d1 = rfl(lds_byte_off);                         // lds_addr
    unsigned g0d2 = rfl((unsigned)(ga & 0xFFFFFFFFu));         // global_addr[31:0]
    unsigned g0d3 = rfl(((unsigned)(ga >> 32) & 0x01FFFFFFu)   // global_addr[56:32]
                        | (2u << 30));                         // type=2 ("image")
    // ---- group 1 ----
    unsigned dim0 = (unsigned)kdim_rem;
    unsigned dim1 = (unsigned)rows_rem;
    unsigned d0 = rfl((1u << 16)        // data_size = 2 bytes
                    | (1u << 20)        // pad_enable
                    | (3u << 22)        // pad_interval: every 16 DWORDs (64B row)
                    | (0u << 25));      // pad_amount: 1 DWORD
    unsigned d1 = rfl((dim0 & 0xFFFFu) << 16);                 // tensor_dim0[15:0]
    unsigned d2 = rfl((dim0 >> 16) | ((dim1 & 0xFFFFu) << 16));// dim0[31:16], dim1[15:0]
    unsigned d3 = rfl((dim1 >> 16) | ((unsigned)BK << 16));    // dim1[31:16], tile_dim0=BK
    unsigned d4 = rfl((unsigned)BM);                           // tile_dim1=BM, tile_dim2=0
    unsigned d5 = rfl((unsigned)kdim_elems);                   // tensor_dim0_stride[31:0]
    unsigned d6 = rfl(0u);                                     // stride0[47:32], stride1 lo
    unsigned d7 = rfl(0u);                                     // stride1 hi
    u32x4 g0 = { g0d0, g0d1, g0d2, g0d3 };
    i32x8 g1 = { (int)d0, (int)d1, (int)d2, (int)d3,
                 (int)d4, (int)d5, (int)d6, (int)d7 };
    i32x4 z4 = { 0, 0, 0, 0 };
    i32x8 z8 = { 0, 0, 0, 0, 0, 0, 0, 0 };
    __builtin_amdgcn_tensor_load_to_lds(g0, g1, z4, z4, z8, 0);
}

// =======================================================================
// 0) zero output
// =======================================================================
__global__ __launch_bounds__(256) void k_zero(float* __restrict__ p, int n4) {
    int i = blockIdx.x * blockDim.x + threadIdx.x;
    if (i < n4) ((float4*)p)[i] = make_float4(0.f, 0.f, 0.f, 0.f);
}

// =======================================================================
// 1) fp32 -> bf16 conversion of x
// =======================================================================
__global__ __launch_bounds__(256) void k_cvt_bf16(const float* __restrict__ src,
                                                  __bf16* __restrict__ dst, int n8) {
    int i = blockIdx.x * blockDim.x + threadIdx.x;
    if (i >= n8) return;
    const float4* s = (const float4*)src + 2 * (size_t)i;
    float4 a = s[0], b = s[1];
    __bf16 o[8] = {(__bf16)a.x, (__bf16)a.y, (__bf16)a.z, (__bf16)a.w,
                   (__bf16)b.x, (__bf16)b.y, (__bf16)b.z, (__bf16)b.w};
    ((uint4*)dst)[i] = *(const uint4*)o;
}

// =======================================================================
// 2) router: scores = sigmoid(rand_logits) (RandomSTE forward), biased top-k,
//    unbiased renorm * 2.5
// =======================================================================
__global__ __launch_bounds__(256) void k_router(const float* __restrict__ rlog,
                                                const float* __restrict__ bias,
                                                int* __restrict__ eid,
                                                float* __restrict__ wgt) {
    int t = blockIdx.x * blockDim.x + threadIdx.x;
    if (t >= T_) return;
    float sc[E_];
#pragma unroll
    for (int e = 0; e < E_; ++e) {
        float x = rlog[(size_t)t * E_ + e];
        sc[e] = 1.f / (1.f + __expf(-x));
    }
    unsigned mask = 0;
    int   ch[K_];
    float cw[K_];
#pragma unroll
    for (int k = 0; k < K_; ++k) {
        float best = -1e30f; int bi = 0;
#pragma unroll
        for (int e = 0; e < E_; ++e) {
            if (mask & (1u << e)) continue;
            float v = sc[e] + bias[e];
            if (v > best) { best = v; bi = e; }   // strict > == lowest-index tie-break
        }
        mask |= 1u << bi; ch[k] = bi; cw[k] = sc[bi];
    }
    float s = cw[0] + cw[1] + cw[2] + cw[3] + 1e-20f;
    float inv = ROUTE_SCALE_ / s;
#pragma unroll
    for (int k = 0; k < K_; ++k) {
        eid[t * K_ + k] = ch[k];
        wgt[t * K_ + k] = cw[k] * inv;
    }
}

// =======================================================================
// 3) deterministic per-expert ranking (stable, matches argsort order),
//    builds slot/tok/weight maps + counts. One block per expert.
// =======================================================================
__global__ __launch_bounds__(256) void k_rank(const int* __restrict__ eid,
                                              const float* __restrict__ wgt,
                                              int* __restrict__ slot_of_a,
                                              int* __restrict__ tok_of_slot,
                                              float* __restrict__ w_of_slot,
                                              int* __restrict__ counts) {
    __shared__ int sc[256];
    int e = blockIdx.x, tid = threadIdx.x;
    int base = 0;
    for (int start = 0; start < TK_; start += 256) {
        int a = start + tid;
        int f = (eid[a] == e) ? 1 : 0;
        sc[tid] = f;
        __syncthreads();
        for (int off = 1; off < 256; off <<= 1) {           // Hillis–Steele inclusive
            int v = (tid >= off) ? sc[tid - off] : 0;
            __syncthreads();
            sc[tid] += v;
            __syncthreads();
        }
        int tot = sc[255];
        if (f) {
            int r = base + sc[tid] - 1;                      // exclusive rank
            if (r < CAP_) {
                int slot = e * CAP_ + r;
                slot_of_a[a]   = slot;
                tok_of_slot[slot] = a / K_;
                w_of_slot[slot]   = wgt[a];
            } else {
                slot_of_a[a] = -1;                           // dropped
            }
        }
        base += tot;
        __syncthreads();
    }
    if (tid == 0) counts[e] = base > CAP_ ? CAP_ : base;
}

// =======================================================================
// 4) dispatch: copy routed token rows (bf16) into [E,CAP,D] buffer
// =======================================================================
__global__ __launch_bounds__(256) void k_dispatch(const __bf16* __restrict__ xbf,
                                                  const int* __restrict__ slot_of_a,
                                                  __bf16* __restrict__ xb) {
    int a = blockIdx.x;
    int s = slot_of_a[a];
    if (s < 0) return;
    int tok = a / K_;
    const uint4* src = (const uint4*)(xbf + (size_t)tok * D_);
    uint4*       dst = (uint4*)(xb + (size_t)s * D_);
    dst[threadIdx.x] = src[threadIdx.x];     // 256 * 16B == 2048 * bf16
}

// =======================================================================
// 5) fused dual-B WMMA GEMM + SwiGLU epilogue:
//    H = silu(A @ Bg) * (A @ Bu), A bf16 [rows,Kdim] staged via TDM,
//    B fp32 [Kdim,Nld] converted+transposed into LDS by the block.
// =======================================================================
__global__ __launch_bounds__(256)
void k_gemm1_swiglu(const __bf16* __restrict__ Aall,
                    const float* __restrict__ Bg_all,
                    const float* __restrict__ Bu_all,
                    __bf16* __restrict__ Hall,
                    int Kdim, int Nld, int rowsTotal,
                    long long aStride, long long bStride, long long hStride,
                    const int* __restrict__ counts) {
    int e = blockIdx.z;
    if (counts && (int)(blockIdx.y * BM) >= counts[e]) return;   // skip empty row-tiles

    const __bf16* A  = Aall  + (size_t)e * aStride;
    const float*  Bg = Bg_all + (size_t)e * bStride;
    const float*  Bu = Bu_all + (size_t)e * bStride;
    __bf16*       H  = Hall  + (size_t)e * hStride;

    const int rowBase = blockIdx.y * BM;
    const int colBase = blockIdx.x * BN;
    const int rowsRem = rowsTotal - rowBase;

    __shared__ __bf16 As[BM * LDA];
    __shared__ __bf16 BgT[BN * LDBT];
    __shared__ __bf16 BuT[BN * LDBT];

    const int tid = threadIdx.x;
    const int lane = tid & 31, wid = tid >> 5;
    const int waveM = (wid & 3) * 32;
    const int waveN = (wid >> 2) * 32;
    const int row16 = lane & 15, half = lane >> 4;
    const unsigned asOff = (unsigned)(uintptr_t)(&As[0]);   // LDS byte offset (low 32b)

    v8f accG[2][2] = {};
    v8f accU[2][2] = {};

    for (int kt = 0; kt < Kdim; kt += BK) {
        __syncthreads();                 // previous iteration's LDS reads complete
        // ---- stage A via Tensor Data Mover (wave 0 issues the DMA) ----
        if (wid == 0) {
            tdm_load_a_tile(asOff, A + (size_t)rowBase * Kdim + kt,
                            Kdim, Kdim - kt, rowsRem);
        }
        // ---- stage B gate/up: fp32 load, bf16 convert, transpose into LDS ----
#pragma unroll
        for (int p = 0; p < 2; ++p) {
            int k  = p * 16 + (tid >> 4);
            int n0 = (tid & 15) * 4;
            const float4 g = *(const float4*)(Bg + (size_t)(kt + k) * Nld + colBase + n0);
            const float4 u = *(const float4*)(Bu + (size_t)(kt + k) * Nld + colBase + n0);
            BgT[(n0 + 0) * LDBT + k] = (__bf16)g.x;
            BgT[(n0 + 1) * LDBT + k] = (__bf16)g.y;
            BgT[(n0 + 2) * LDBT + k] = (__bf16)g.z;
            BgT[(n0 + 3) * LDBT + k] = (__bf16)g.w;
            BuT[(n0 + 0) * LDBT + k] = (__bf16)u.x;
            BuT[(n0 + 1) * LDBT + k] = (__bf16)u.y;
            BuT[(n0 + 2) * LDBT + k] = (__bf16)u.z;
            BuT[(n0 + 3) * LDBT + k] = (__bf16)u.w;
        }
        if (wid == 0) __builtin_amdgcn_s_wait_tensorcnt(0);   // TDM tile landed
        __syncthreads();
        if (kt + BK < Kdim) {           // gfx1250 global_prefetch_b8 on next tiles
            __builtin_prefetch(Bg + (size_t)(kt + BK) * Nld + colBase, 0, 1);
            __builtin_prefetch(Bu + (size_t)(kt + BK) * Nld + colBase, 0, 1);
        }
        // ---- fragments per ISA 7.12.2 layouts ----
        AFrag a[2];
#pragma unroll
        for (int mi = 0; mi < 2; ++mi) {
            int r = waveM + mi * 16 + row16;
#pragma unroll
            for (int v = 0; v < 8; ++v) {
                int k = 2 * v + ((v >= 4) ? 8 : 0) + half * 8;
                a[mi].u[v] = *(const unsigned*)&As[r * LDA + k];
            }
        }
        BFrag bg[2], bu[2];
#pragma unroll
        for (int ni = 0; ni < 2; ++ni) {
            int c = waveN + ni * 16 + row16;
#pragma unroll
            for (int v = 0; v < 8; ++v) {
                int k = 2 * v + ((v >= 4) ? 8 : 0) + half * 8;
                bg[ni].u[v] = *(const unsigned*)&BgT[c * LDBT + k];
                bu[ni].u[v] = *(const unsigned*)&BuT[c * LDBT + k];
            }
        }
#pragma unroll
        for (int mi = 0; mi < 2; ++mi)
#pragma unroll
            for (int ni = 0; ni < 2; ++ni) {
                accG[mi][ni] = wmma_bf16(a[mi], bg[ni], accG[mi][ni]);
                accU[mi][ni] = wmma_bf16(a[mi], bu[ni], accU[mi][ni]);
            }
    }
    // ---- SwiGLU epilogue, bf16 store ----
#pragma unroll
    for (int mi = 0; mi < 2; ++mi)
#pragma unroll
        for (int ni = 0; ni < 2; ++ni) {
            int gc = colBase + waveN + ni * 16 + row16;
#pragma unroll
            for (int r = 0; r < 8; ++r) {
                int gr = rowBase + waveM + mi * 16 + half * 8 + r;
                float g = accG[mi][ni][r];
                float u = accU[mi][ni][r];
                float h = (g / (1.f + __expf(-g))) * u;
                H[(size_t)gr * Nld + gc] = (__bf16)h;
            }
        }
}

// =======================================================================
// 6) expert down-proj WMMA GEMM + weighted scatter-combine epilogue
//    Y = H @ Wd ; out[tok] += w * Y[slot]   (global_atomic_add_f32)
// =======================================================================
__global__ __launch_bounds__(256)
void k_gemm2_combine(const __bf16* __restrict__ Hall,
                     const float* __restrict__ Wd_all,
                     float* __restrict__ out,
                     const int* __restrict__ counts,
                     const int* __restrict__ tok_of_slot,
                     const float* __restrict__ w_of_slot) {
    const int e = blockIdx.z;
    const int cnt = counts[e];
    if ((int)(blockIdx.y * BM) >= cnt) return;

    const __bf16* A = Hall   + (size_t)e * CAP_ * M_;
    const float*  B = Wd_all + (size_t)e * M_ * D_;

    const int rowBase = blockIdx.y * BM;
    const int colBase = blockIdx.x * BN;
    const int rowsRem = CAP_ - rowBase;

    __shared__ __bf16 As[BM * LDA];
    __shared__ __bf16 BT[BN * LDBT];

    const int tid = threadIdx.x;
    const int lane = tid & 31, wid = tid >> 5;
    const int waveM = (wid & 3) * 32;
    const int waveN = (wid >> 2) * 32;
    const int row16 = lane & 15, half = lane >> 4;
    const unsigned asOff = (unsigned)(uintptr_t)(&As[0]);

    v8f acc[2][2] = {};

    for (int kt = 0; kt < M_; kt += BK) {
        __syncthreads();
        if (wid == 0) {
            tdm_load_a_tile(asOff, A + (size_t)rowBase * M_ + kt,
                            M_, M_ - kt, rowsRem);
        }
#pragma unroll
        for (int p = 0; p < 2; ++p) {
            int k  = p * 16 + (tid >> 4);
            int n0 = (tid & 15) * 4;
            const float4 b = *(const float4*)(B + (size_t)(kt + k) * D_ + colBase + n0);
            BT[(n0 + 0) * LDBT + k] = (__bf16)b.x;
            BT[(n0 + 1) * LDBT + k] = (__bf16)b.y;
            BT[(n0 + 2) * LDBT + k] = (__bf16)b.z;
            BT[(n0 + 3) * LDBT + k] = (__bf16)b.w;
        }
        if (wid == 0) __builtin_amdgcn_s_wait_tensorcnt(0);
        __syncthreads();
        if (kt + BK < M_)
            __builtin_prefetch(B + (size_t)(kt + BK) * D_ + colBase, 0, 1);

        AFrag a[2];
#pragma unroll
        for (int mi = 0; mi < 2; ++mi) {
            int r = waveM + mi * 16 + row16;
#pragma unroll
            for (int v = 0; v < 8; ++v) {
                int k = 2 * v + ((v >= 4) ? 8 : 0) + half * 8;
                a[mi].u[v] = *(const unsigned*)&As[r * LDA + k];
            }
        }
        BFrag bf[2];
#pragma unroll
        for (int ni = 0; ni < 2; ++ni) {
            int c = waveN + ni * 16 + row16;
#pragma unroll
            for (int v = 0; v < 8; ++v) {
                int k = 2 * v + ((v >= 4) ? 8 : 0) + half * 8;
                bf[ni].u[v] = *(const unsigned*)&BT[c * LDBT + k];
            }
        }
#pragma unroll
        for (int mi = 0; mi < 2; ++mi)
#pragma unroll
            for (int ni = 0; ni < 2; ++ni)
                acc[mi][ni] = wmma_bf16(a[mi], bf[ni], acc[mi][ni]);
    }
    // ---- weighted scatter-add combine ----
#pragma unroll
    for (int mi = 0; mi < 2; ++mi)
#pragma unroll
        for (int ni = 0; ni < 2; ++ni) {
            int gc = colBase + waveN + ni * 16 + row16;
#pragma unroll
            for (int r = 0; r < 8; ++r) {
                int rowInExp = rowBase + waveM + mi * 16 + half * 8 + r;
                if (rowInExp < cnt) {
                    int slot = e * CAP_ + rowInExp;
                    int tok  = tok_of_slot[slot];
                    float w  = w_of_slot[slot];
                    __hip_atomic_fetch_add(&out[(size_t)tok * D_ + gc],
                                           w * acc[mi][ni][r],
                                           __ATOMIC_RELAXED, __HIP_MEMORY_SCOPE_AGENT);
                }
            }
        }
}

// =======================================================================
// 7) shared-expert down-proj: out += Hs @ SWd (dense, plain add; runs last)
// =======================================================================
__global__ __launch_bounds__(256)
void k_gemm2_shared(const __bf16* __restrict__ Hs,
                    const float* __restrict__ Wd,
                    float* __restrict__ out) {
    const int rowBase = blockIdx.y * BM;
    const int colBase = blockIdx.x * BN;
    const int rowsRem = T_ - rowBase;

    __shared__ __bf16 As[BM * LDA];
    __shared__ __bf16 BT[BN * LDBT];

    const int tid = threadIdx.x;
    const int lane = tid & 31, wid = tid >> 5;
    const int waveM = (wid & 3) * 32;
    const int waveN = (wid >> 2) * 32;
    const int row16 = lane & 15, half = lane >> 4;
    const unsigned asOff = (unsigned)(uintptr_t)(&As[0]);

    v8f acc[2][2] = {};

    for (int kt = 0; kt < MS_; kt += BK) {
        __syncthreads();
        if (wid == 0) {
            tdm_load_a_tile(asOff, Hs + (size_t)rowBase * MS_ + kt,
                            MS_, MS_ - kt, rowsRem);
        }
#pragma unroll
        for (int p = 0; p < 2; ++p) {
            int k  = p * 16 + (tid >> 4);
            int n0 = (tid & 15) * 4;
            const float4 b = *(const float4*)(Wd + (size_t)(kt + k) * D_ + colBase + n0);
            BT[(n0 + 0) * LDBT + k] = (__bf16)b.x;
            BT[(n0 + 1) * LDBT + k] = (__bf16)b.y;
            BT[(n0 + 2) * LDBT + k] = (__bf16)b.z;
            BT[(n0 + 3) * LDBT + k] = (__bf16)b.w;
        }
        if (wid == 0) __builtin_amdgcn_s_wait_tensorcnt(0);
        __syncthreads();
        if (kt + BK < MS_)
            __builtin_prefetch(Wd + (size_t)(kt + BK) * D_ + colBase, 0, 1);

        AFrag a[2];
#pragma unroll
        for (int mi = 0; mi < 2; ++mi) {
            int r = waveM + mi * 16 + row16;
#pragma unroll
            for (int v = 0; v < 8; ++v) {
                int k = 2 * v + ((v >= 4) ? 8 : 0) + half * 8;
                a[mi].u[v] = *(const unsigned*)&As[r * LDA + k];
            }
        }
        BFrag bf[2];
#pragma unroll
        for (int ni = 0; ni < 2; ++ni) {
            int c = waveN + ni * 16 + row16;
#pragma unroll
            for (int v = 0; v < 8; ++v) {
                int k = 2 * v + ((v >= 4) ? 8 : 0) + half * 8;
                bf[ni].u[v] = *(const unsigned*)&BT[c * LDBT + k];
            }
        }
#pragma unroll
        for (int mi = 0; mi < 2; ++mi)
#pragma unroll
            for (int ni = 0; ni < 2; ++ni)
                acc[mi][ni] = wmma_bf16(a[mi], bf[ni], acc[mi][ni]);
    }
#pragma unroll
    for (int mi = 0; mi < 2; ++mi)
#pragma unroll
        for (int ni = 0; ni < 2; ++ni) {
            int gc = colBase + waveN + ni * 16 + row16;
#pragma unroll
            for (int r = 0; r < 8; ++r) {
                int gr = rowBase + waveM + mi * 16 + half * 8 + r;
                out[(size_t)gr * D_ + gc] += acc[mi][ni][r];
            }
        }
}

// =======================================================================
// launcher
// =======================================================================
extern "C" void kernel_launch(void* const* d_in, const int* in_sizes, int n_in,
                              void* d_out, int out_size, void* d_ws, size_t ws_size,
                              hipStream_t stream) {
    const float* x        = (const float*)d_in[0];
    const float* rlog     = (const float*)d_in[1];
    const float* ebias    = (const float*)d_in[2];
    // d_in[3] = gate_w: unused in forward (RandomSTE forward == rand_logits)
    const float* w_gate   = (const float*)d_in[4];
    const float* w_up     = (const float*)d_in[5];
    const float* w_down   = (const float*)d_in[6];
    const float* sw_gate  = (const float*)d_in[7];
    const float* sw_up    = (const float*)d_in[8];
    const float* sw_down  = (const float*)d_in[9];
    float* out = (float*)d_out;

    // ---- workspace carve-up (256B aligned) ----
    char* p = (char*)d_ws;
    auto alloc = [&](size_t bytes) -> void* {
        void* r = (void*)p;
        p += (bytes + 255) & ~(size_t)255;
        return r;
    };
    __bf16* xbf   = (__bf16*)alloc((size_t)T_ * D_ * 2);          //  8 MB
    __bf16* xb    = (__bf16*)alloc((size_t)E_ * CAP_ * D_ * 2);   // 64 MB
    __bf16* Hbuf  = (__bf16*)alloc((size_t)E_ * CAP_ * M_ * 2);   // 44 MB
    __bf16* Hs    = (__bf16*)alloc((size_t)T_ * MS_ * 2);         // 11 MB
    int*    eid   = (int*)  alloc((size_t)TK_ * 4);
    float*  wgt   = (float*)alloc((size_t)TK_ * 4);
    int*    slotA = (int*)  alloc((size_t)TK_ * 4);
    int*    tokS  = (int*)  alloc((size_t)E_ * CAP_ * 4);
    float*  wS    = (float*)alloc((size_t)E_ * CAP_ * 4);
    int*    cnts  = (int*)  alloc((size_t)E_ * 4);

    // 0) out = 0
    {
        int n4 = T_ * D_ / 4;
        k_zero<<<(n4 + 255) / 256, 256, 0, stream>>>(out, n4);
    }
    // 1) x -> bf16
    {
        int n8 = T_ * D_ / 8;
        k_cvt_bf16<<<(n8 + 255) / 256, 256, 0, stream>>>(x, xbf, n8);
    }
    // 2) router
    k_router<<<(T_ + 255) / 256, 256, 0, stream>>>(rlog, ebias, eid, wgt);
    // 3) rank / slot maps (one block per expert, deterministic)
    k_rank<<<E_, 256, 0, stream>>>(eid, wgt, slotA, tokS, wS, cnts);
    // 4) dispatch token rows
    k_dispatch<<<TK_, 256, 0, stream>>>(xbf, slotA, xb);
    // 5) expert SwiGLU up/gate (count-gated tiles)
    {
        dim3 g(M_ / BN, CAP_ / BM, E_);
        k_gemm1_swiglu<<<g, 256, 0, stream>>>(
            xb, w_gate, w_up, Hbuf, D_, M_, CAP_,
            (long long)CAP_ * D_, (long long)D_ * M_, (long long)CAP_ * M_, cnts);
    }
    // 6) expert down-proj + weighted scatter combine
    {
        dim3 g(D_ / BN, CAP_ / BM, E_);
        k_gemm2_combine<<<g, 256, 0, stream>>>(Hbuf, w_down, out, cnts, tokS, wS);
    }
    // 7) shared expert SwiGLU
    {
        dim3 g(MS_ / BN, T_ / BM, 1);
        k_gemm1_swiglu<<<g, 256, 0, stream>>>(
            xbf, sw_gate, sw_up, Hs, D_, MS_, T_, 0, 0, 0, nullptr);
    }
    // 8) shared expert down-proj, plain add (after MoE combine in stream order)
    {
        dim3 g(D_ / BN, T_ / BM, 1);
        k_gemm2_shared<<<g, 256, 0, stream>>>(Hs, sw_down, out);
    }
}
